// BiLSTMEncoder_37331855737091
// MI455X (gfx1250) — compile-verified, matrix operand order checked by reading, not run on hardware
//
#include <hip/hip_runtime.h>

// ---------------------------------------------------------------------------
// BiLSTM encoder for MI455X (gfx1250, wave32, WMMA, 320KB LDS/WGP)
// B=32, T=1024, D=H=512, L=2, gates=4H=2048, K=D+H=1024
// ---------------------------------------------------------------------------

#define B_        32
#define T_        1024
#define D_        512
#define H_        512
#define G4H_      2048
#define KTOT_     1024
#define NT_       128      // N tiles of 16
#define WGPD_     16       // workgroups per direction in recurrent kernel
#define THREADS_  256

typedef __attribute__((ext_vector_type(16))) __bf16 v16bf;
typedef __attribute__((ext_vector_type(8)))  float  v8f;

static __device__ __forceinline__ unsigned short f2bf(float f) {
  unsigned u = __float_as_uint(f);
  u += 0x7FFFu + ((u >> 16) & 1u);          // round-to-nearest-even
  return (unsigned short)(u >> 16);
}
static __device__ __forceinline__ float sigm(float x) {
  return 1.0f / (1.0f + __expf(-x));
}

// ---- fragment index helpers -----------------------------------------------
// A-fragment (16x32 bf16, M x K): lane = (m&15) | (khalf<<4), 16 bf16/lane.
// B-fragment (32x16 bf16, K x N): lane = (n&15) | (khalf<<4), 16 bf16/lane.
// khalf = bit3 of k within 32-block; element e = (k&7) | (bit4(k)<<3).
static __device__ __forceinline__ size_t wpk_idx(int dl, int nt, int k, int n) {
  int ks   = k >> 5;
  int lane = (n & 15) | (((k >> 3) & 1) << 4);
  int e    = (k & 7)  | (((k >> 4) & 1) << 3);
  return ((((size_t)dl * NT_ + nt) * 32 + ks) * 512) + (size_t)lane * 16 + e;
}
static __device__ __forceinline__ size_t afrag_idx(int dl, int t, int m, int k) {
  int mt   = m >> 4;
  int ks   = k >> 5;
  int lane = (m & 15) | (((k >> 3) & 1) << 4);
  int e    = (k & 7)  | (((k >> 4) & 1) << 3);
  return ((((size_t)dl * T_ + t) * 2 + mt) * 32 + ks) * 512 + (size_t)lane * 16 + e;
}

// ---- grid barrier (per-direction, monotonic counter) ----------------------
static __device__ __forceinline__ void gbar(unsigned* ctr, unsigned target) {
  __syncthreads();
  if (threadIdx.x == 0) {
    __threadfence();
    atomicAdd(ctr, 1u);
    volatile unsigned* vc = (volatile unsigned*)ctr;
    while (*vc < target) __builtin_amdgcn_s_sleep(1);
    __threadfence();
  }
  __syncthreads();
}

// ---------------------------------------------------------------------------
// init: zero barrier counters + zero h-part of t=0 A-fragments (all dir/layer)
// ---------------------------------------------------------------------------
__global__ void init_kernel(unsigned short* afrag, unsigned* bar) {
  size_t i = (size_t)blockIdx.x * blockDim.x + threadIdx.x;
  if (i < 2) bar[i] = 0u;
  if (i < 65536) {
    int dl  = (int)(i >> 14);
    int r   = (int)(i & 16383);
    int mt  = r >> 13;
    int r2  = r & 8191;
    int ks  = 16 + (r2 >> 9);     // h-part ksteps 16..31
    int off = r2 & 511;
    size_t idx = ((((size_t)dl * T_ + 0) * 2 + mt) * 32 + ks) * 512 + off;
    afrag[idx] = 0;
  }
}

// ---------------------------------------------------------------------------
// pack weights (f32 -> bf16 B-fragment layout), 4 x (1024 x 2048)
// ---------------------------------------------------------------------------
__global__ void pack_weights_kernel(const float* __restrict__ Wf,
                                    const float* __restrict__ Wb,
                                    unsigned short* __restrict__ wpk) {
  size_t i = (size_t)blockIdx.x * blockDim.x + threadIdx.x;
  const size_t per = (size_t)KTOT_ * G4H_;
  if (i >= 4 * per) return;
  int dl = (int)(i / per);
  size_t r = i % per;
  int k = (int)(r / G4H_);
  int n = (int)(r % G4H_);
  int dir = dl >> 1, layer = dl & 1;
  const float* W = (dir == 0 ? Wf : Wb) + (size_t)layer * per;
  wpk[wpk_idx(dl, n >> 4, k, n)] = f2bf(W[(size_t)k * G4H_ + n]);
}

// ---------------------------------------------------------------------------
// pack inputs: X (fwd) and length-reversed X (bwd) -> layer-0 A-frag X-part
// ---------------------------------------------------------------------------
__global__ void pack_inputs_kernel(const float* __restrict__ X,
                                   const int* __restrict__ len,
                                   unsigned short* __restrict__ afrag) {
  size_t i = (size_t)blockIdx.x * blockDim.x + threadIdx.x;
  const size_t per = (size_t)B_ * T_ * D_;
  if (i >= 2 * per) return;
  int dir = (int)(i / per);
  size_t r = i % per;
  int b  = (int)(r / ((size_t)T_ * D_));
  size_t r2 = r % ((size_t)T_ * D_);
  int t  = (int)(r2 / D_);
  int kx = (int)(r2 % D_);
  int src_t = t;
  if (dir == 1) { int ln = len[b]; src_t = (t < ln) ? (ln - 1 - t) : t; }
  float v = X[((size_t)b * T_ + src_t) * D_ + kx];
  afrag[afrag_idx(dir * 2 + 0, t, b, kx)] = f2bf(v);
}

// ---------------------------------------------------------------------------
// persistent recurrent kernel: 32 WGs = 2 dirs x 16 WGs, 256 thr (8 waves)
// Weights live in LDS (8 N-tiles x 32KB = 256KB) for the whole layer.
// ---------------------------------------------------------------------------
__launch_bounds__(THREADS_)
__global__ void bilstm_recurrent_kernel(unsigned short* __restrict__ afrag,
                                        const unsigned short* __restrict__ wpk,
                                        float* __restrict__ zbuf,
                                        float* __restrict__ hout,
                                        unsigned* __restrict__ bar,
                                        const int* __restrict__ len,
                                        const float* __restrict__ bf_p,
                                        const float* __restrict__ bb_p) {
  extern __shared__ unsigned short smem[];   // 256 KB bf16 weight fragments
  const int dir  = blockIdx.x / WGPD_;
  const int wg   = blockIdx.x % WGPD_;
  const int wave = threadIdx.x >> 5;
  const int lane = threadIdx.x & 31;
  const int nt   = wg * 8 + wave;            // this wave's N-tile (0..127)
  float*    zd   = zbuf + (size_t)dir * B_ * G4H_;
  unsigned* ctr  = bar + dir;
  unsigned  ep   = 0;

  // fixed element ownership for the gate phase (c/h state in registers)
  const int tid_d = wg * THREADS_ + threadIdx.x;   // 0..4095 per dir
  const int pb    = tid_d >> 7;                    // batch row 0..31
  const int pn4   = (tid_d & 127) << 2;            // 4 H-columns
  const int ln    = len[pb];

  for (int layer = 0; layer < 2; ++layer) {
    const int dl = dir * 2 + layer;
    // stage this wave's weight tile block into LDS (32 KB / wave)
    {
      const uint4* src = (const uint4*)(wpk + (((size_t)dl * NT_ + nt) * 32) * 512);
      uint4*       dst = (uint4*)(smem + (size_t)wave * 16384);
      for (int i = lane; i < 2048; i += 32) dst[i] = src[i];
    }
    __syncthreads();
    const float* bias = (dir == 0 ? bf_p : bb_p) + (size_t)layer * G4H_;
    float cst[4] = {0.f, 0.f, 0.f, 0.f};
    float hst[4] = {0.f, 0.f, 0.f, 0.f};

    for (int t = 0; t < T_; ++t) {
      // ---------------- phase 1: z = [x,h] @ W  (two 16x16 tiles, same nt)
      const unsigned short* Ab = afrag + ((size_t)dl * T_ + t) * 32768;
      v8f acc0 = {};
      v8f acc1 = {};
#pragma unroll 4
      for (int ks = 0; ks < 32; ++ks) {
        v16bf bfr = *(const v16bf*)(smem + (size_t)wave * 16384 + ks * 512 + lane * 16);
        v16bf a0  = *(const v16bf*)(Ab + ks * 512 + lane * 16);
        v16bf a1  = *(const v16bf*)(Ab + 16384 + ks * 512 + lane * 16);
        acc0 = __builtin_amdgcn_wmma_f32_16x16x32_bf16(false, a0, false, bfr,
                                                       (short)0, acc0, false, false);
        acc1 = __builtin_amdgcn_wmma_f32_16x16x32_bf16(false, a1, false, bfr,
                                                       (short)0, acc1, false, false);
      }
      {
        const int nn = nt * 16 + (lane & 15);
        const int mr = (lane >> 4) * 8;            // C layout: M = r + 8*(lane>=16)
#pragma unroll
        for (int r = 0; r < 8; ++r) {
          zd[(size_t)(mr + r) * G4H_ + nn]        = acc0[r];
          zd[(size_t)(16 + mr + r) * G4H_ + nn]   = acc1[r];
        }
      }
      ep++; gbar(ctr, ep * WGPD_);

      // ---------------- phase 3: gates + state update + h broadcast
#pragma unroll
      for (int j = 0; j < 4; ++j) {
        const int n = pn4 + j;
        float zi = zd[(size_t)pb * G4H_ + n]          + bias[n];
        float zg = zd[(size_t)pb * G4H_ + 512 + n]    + bias[512 + n];
        float zf = zd[(size_t)pb * G4H_ + 1024 + n]   + bias[1024 + n];
        float zo = zd[(size_t)pb * G4H_ + 1536 + n]   + bias[1536 + n];
        float cn = sigm(zf + 1.0f) * cst[j] + sigm(zi) * tanhf(zg);
        float hn = sigm(zo) * tanhf(cn);
        bool  mk = (t < ln);
        float ck = mk ? cn : cst[j];
        float hk = mk ? hn : hst[j];
        float y  = mk ? hn : 0.0f;
        cst[j] = ck;
        hst[j] = hk;
        if (t + 1 < T_)
          afrag[afrag_idx(dl, t + 1, pb, 512 + n)] = f2bf(hk);   // h -> next step
        if (layer == 0)
          afrag[afrag_idx(dir * 2 + 1, t, pb, n)] = f2bf(y);     // y -> layer-2 X
        else
          hout[(size_t)dir * B_ * T_ * H_ + ((size_t)pb * T_ + t) * H_ + n] = y;
      }
      ep++; gbar(ctr, ep * WGPD_);
    }
    __syncthreads();
  }
}

// ---------------------------------------------------------------------------
// finalize: out[:, :, :H] = fwd ; out[:, :, H:] = length-reversed bwd
// ---------------------------------------------------------------------------
__global__ void finalize_kernel(const float* __restrict__ hout,
                                const int* __restrict__ len,
                                float* __restrict__ out) {
  size_t i = (size_t)blockIdx.x * blockDim.x + threadIdx.x;
  const size_t total = (size_t)B_ * T_ * (2 * H_);
  if (i >= total) return;
  int b = (int)(i / ((size_t)T_ * 2 * H_));
  size_t r = i % ((size_t)T_ * 2 * H_);
  int t = (int)(r / (2 * H_));
  int n = (int)(r % (2 * H_));
  float v;
  if (n < H_) {
    v = hout[((size_t)b * T_ + t) * H_ + n];
  } else {
    int lnv = len[b];
    int st  = (t < lnv) ? (lnv - 1 - t) : t;
    v = hout[(size_t)B_ * T_ * H_ + ((size_t)b * T_ + st) * H_ + (n - H_)];
  }
  out[i] = v;
}

// ---------------------------------------------------------------------------
extern "C" void kernel_launch(void* const* d_in, const int* in_sizes, int n_in,
                              void* d_out, int out_size, void* d_ws, size_t ws_size,
                              hipStream_t stream) {
  (void)in_sizes; (void)n_in; (void)out_size; (void)ws_size;
  const float* X   = (const float*)d_in[0];
  const int*   len = (const int*)d_in[1];
  const float* Wf  = (const float*)d_in[2];
  const float* bf  = (const float*)d_in[3];
  const float* Wb  = (const float*)d_in[4];
  const float* bb  = (const float*)d_in[5];
  float* out = (float*)d_out;

  // workspace carve-up (all 256B-aligned by construction)
  char* p = (char*)d_ws;
  unsigned short* wpk = (unsigned short*)p;            // 4 * 2,097,152 bf16 = 16 MiB
  p += (size_t)4 * 2097152 * 2;
  unsigned short* afrag = (unsigned short*)p;          // 4 * 1024 * 32768 bf16 = 256 MiB
  p += (size_t)4 * 1024 * 32768 * 2;
  float* zbuf = (float*)p;                             // 2 * 32 * 2048 f32 = 512 KiB
  p += (size_t)2 * 32 * 2048 * 4;
  float* hout = (float*)p;                             // 2 * B*T*H f32 = 128 MiB
  p += (size_t)2 * B_ * T_ * H_ * 4;
  unsigned* bar = (unsigned*)p;                        // 2 counters

  (void)hipFuncSetAttribute((const void*)bilstm_recurrent_kernel,
                            hipFuncAttributeMaxDynamicSharedMemorySize, 262144);

  init_kernel<<<256, 256, 0, stream>>>(afrag, bar);
  pack_weights_kernel<<<32768, 256, 0, stream>>>(Wf, Wb, wpk);
  pack_inputs_kernel<<<131072, 256, 0, stream>>>(X, len, afrag);
  bilstm_recurrent_kernel<<<2 * WGPD_, THREADS_, 262144, stream>>>(
      afrag, wpk, zbuf, hout, bar, len, bf, bb);
  finalize_kernel<<<131072, 256, 0, stream>>>(hout, len, out);
}
